// HiddenLayer_68212670595564
// MI455X (gfx1250) — matrix operand order repaired and checked by reading, not compiled, not run on hardware
//
#include <hip/hip_runtime.h>
#include <hip/hip_bf16.h>
#include <cmath>

#define H_   200
#define D_   24
#define B_   64
#define T_   2048
#define GH   800      // 4*H
#define AST  232      // LDS A-tile row stride (bf16); K padded to 224=7*32
#define NT_G 50       // gate N-tiles (800/16)
#define NT_V 13       // ve N-tiles (ceil(200/16))
#define KC   7        // K chunks of 32 (224)

typedef __attribute__((ext_vector_type(16))) __bf16 v16bf;
typedef __attribute__((ext_vector_type(8)))  __bf16 v8bf;
typedef __attribute__((ext_vector_type(4)))  __bf16 v4bf;
typedef __attribute__((ext_vector_type(8)))  float  v8f;

// ---------------------------------------------------------------------------
// Prep: repack U_w / W_w / V_w (fp32) into bf16 B-operand fragments in the
// contiguous-k order this file uses for both v_wmma_f32_16x16x32_bf16 operands:
//   frag element b[i] = B[k = 32*kc + 16*(lane>>4) + i][n = lane&15]
// with gate permutation g' = 4*h' + q  <->  old row q*H + h', so each 16-wide
// N tile = 4 hidden units x {i,f,g,o} (divides 800 evenly, and the cell update
// reads i,f,g,o as one contiguous float4).
// ---------------------------------------------------------------------------
__global__ void prep_kernel(const float* __restrict__ Uw, const float* __restrict__ Ub,
                            const float* __restrict__ Ww, const float* __restrict__ Vw,
                            __bf16* __restrict__ wsU, __bf16* __restrict__ wsW,
                            __bf16* __restrict__ wsV, float* __restrict__ wsUb) {
    const int NU = NT_G * KC * 32 * 16;   // 179200 elements per 800x224 matrix
    int idx = blockIdx.x * blockDim.x + threadIdx.x;
    if (idx < 2 * NU) {
        const float* src = (idx < NU) ? Uw : Ww;
        __bf16*      dst = (idx < NU) ? wsU : wsW;
        int e    = (idx < NU) ? idx : idx - NU;
        int i    = e & 15;
        int lane = (e >> 4) & 31;
        int r    = e >> 9;
        int kc   = r % KC;
        int tile = r / KC;
        int n    = lane & 15;
        int kh   = lane >> 4;
        int gp   = tile * 16 + n;              // permuted gate index
        int k    = kc * 32 + kh * 16 + i;
        int go   = (gp & 3) * H_ + (gp >> 2);  // original gate row
        float v  = (k < H_) ? src[go * H_ + k] : 0.0f;
        dst[e] = (__bf16)v;
    } else if (idx < 2 * NU + NT_V * 512) {
        int e    = idx - 2 * NU;
        int i    = e & 15;
        int lane = (e >> 4) & 31;
        int tile = e >> 9;                     // kc always 0 (D=24 < 32)
        int n    = lane & 15;
        int kh   = lane >> 4;
        int hp   = tile * 16 + n;
        int k    = kh * 16 + i;
        float v  = (hp < H_ && k < D_) ? Vw[hp * D_ + k] : 0.0f;
        wsV[e] = (__bf16)v;
    } else if (idx < 2 * NU + NT_V * 512 + GH) {
        int gp = idx - (2 * NU + NT_V * 512);
        int go = (gp & 3) * H_ + (gp >> 2);
        wsUb[gp] = Ub[go];
    }
}

// ---------------------------------------------------------------------------
// Fused LSTM scan: 4 blocks x 256 threads (8 waves). Block bb owns batch rows
// [16*bb, 16*bb+16) for all T. Software-pipelined: step t+1's h_past/emb rows
// are prefetched into registers while step t's WMMAs run, hiding HBM latency
// on the serial chain. 2 barriers/step.
// ---------------------------------------------------------------------------
__global__ __launch_bounds__(256, 1)
void lstm_scan_kernel(const float* __restrict__ emb, const float* __restrict__ hpast,
                      const __bf16* __restrict__ wsU, const __bf16* __restrict__ wsW,
                      const __bf16* __restrict__ wsV, const float* __restrict__ wsUb,
                      const float* __restrict__ Vb, const float* __restrict__ h0,
                      const float* __restrict__ c0, float* __restrict__ out) {
    __shared__ __bf16 Au[16 * AST];     // h_past_t tile (A operand, K padded)
    __shared__ __bf16 Ah[16 * AST];     // current h tile (A operand)
    __shared__ __bf16 Ae[16 * 32];      // emb_t tile (A operand, D padded to 32)
    __shared__ float  gates[16 * GH];   // pre-activation gates, interleaved i,f,g,o
    __shared__ float  vebuf[16 * 208];  // V(e_t) + V_b pre-activation
    __shared__ float  cbuf[16 * H_];    // cell state

    const int tid  = threadIdx.x;
    const int b0   = blockIdx.x * 16;
    const int lane = tid & 31;
    const int wave = __builtin_amdgcn_readfirstlane(tid >> 5);  // SGPR -> scalar tile loop
    const int n    = lane & 15;
    const int kh   = lane >> 4;

    // ---- one-time init: zero K-padding, load h0/c0 ----
    for (int idx = tid; idx < 16 * AST; idx += 256) { Au[idx] = (__bf16)0.0f; Ah[idx] = (__bf16)0.0f; }
    for (int idx = tid; idx < 16 * 32;  idx += 256) Ae[idx] = (__bf16)0.0f;
    for (int idx = tid; idx < 16 * H_;  idx += 256) {
        int r = idx / H_, c = idx - r * H_;
        Ah[r * AST + c] = (__bf16)h0[c];
        cbuf[idx] = c0[c];
    }

    // ---- register prefetch buffers (float4-granular staging) ----
    float4 ru[4];   // 800 float4 chunks of h_past rows / 256 threads
    float4 re;      // 96 float4 chunks of emb rows (tid < 96)

    auto prefetch = [&](int t) {
#pragma unroll
        for (int j = 0; j < 4; ++j) {
            int idx = tid + j * 256;
            if (idx < 16 * 50) {
                int r = idx / 50, c = idx - r * 50;
                ru[j] = *(const float4*)&hpast[((size_t)(b0 + r) * T_ + t) * H_ + c * 4];
            }
        }
        if (tid < 16 * 6) {
            int r = tid / 6, c = tid - r * 6;
            re = *(const float4*)&emb[((size_t)(b0 + r) * T_ + t) * D_ + c * 4];
        }
    };
    auto stage = [&]() {
#pragma unroll
        for (int j = 0; j < 4; ++j) {
            int idx = tid + j * 256;
            if (idx < 16 * 50) {
                int r = idx / 50, c = idx - r * 50;
                v4bf p; p[0] = (__bf16)ru[j].x; p[1] = (__bf16)ru[j].y;
                        p[2] = (__bf16)ru[j].z; p[3] = (__bf16)ru[j].w;
                *(v4bf*)&Au[r * AST + c * 4] = p;
            }
        }
        if (tid < 16 * 6) {
            int r = tid / 6, c = tid - r * 6;
            v4bf p; p[0] = (__bf16)re.x; p[1] = (__bf16)re.y;
                    p[2] = (__bf16)re.z; p[3] = (__bf16)re.w;
            *(v4bf*)&Ae[r * 32 + c * 4] = p;
        }
    };

    prefetch(0);
    __syncthreads();            // init (incl. LDS pad zeroing) complete

    for (int t = 0; t < T_; ++t) {
        // -------- stage step-t inputs from regs, then make visible --------
        stage();
        __syncthreads();        // B1: Au/Ae(t) + Ah (from t-1 cell update) visible

        if (t + 1 < T_) prefetch(t + 1);   // overlap HBM latency with GEMMs below

        // -------- GEMM phase: hoist A fragments once per wave --------
        // frag a[i]: row m = n; k = 32kc+16*kh+i
        v16bf au[KC], ah[KC];
#pragma unroll
        for (int kc = 0; kc < KC; ++kc) {
            const __bf16* pu = &Au[n * AST + kc * 32 + kh * 16];
            const __bf16* ph = &Ah[n * AST + kc * 32 + kh * 16];
            v8bf lu = *(const v8bf*)pu, hu = *(const v8bf*)(pu + 8);
            v8bf lh = *(const v8bf*)ph, hh = *(const v8bf*)(ph + 8);
#pragma unroll
            for (int i = 0; i < 8; ++i) {
                au[kc][i] = lu[i]; au[kc][8 + i] = hu[i];
                ah[kc][i] = lh[i]; ah[kc][8 + i] = hh[i];
            }
        }
        v16bf ae;
        {
            const __bf16* pe = &Ae[n * 32 + kh * 16];
            v8bf le = *(const v8bf*)pe, he = *(const v8bf*)(pe + 8);
#pragma unroll
            for (int i = 0; i < 8; ++i) { ae[i] = le[i]; ae[8 + i] = he[i]; }
        }

        for (int tile = wave; tile < NT_G + NT_V; tile += 8) {   // uniform (wave is SGPR)
            if (tile < NT_G) {
                // Two independent accumulator chains (U-path, W-path) for ILP.
                v8f accu = {}, accw = {};
#pragma unroll
                for (int kc = 0; kc < KC; ++kc) {
                    v16bf bu = *(const v16bf*)(wsU + ((size_t)(tile * KC + kc) * 32 + lane) * 16);
                    v16bf bw = *(const v16bf*)(wsW + ((size_t)(tile * KC + kc) * 32 + lane) * 16);
                    accu = __builtin_amdgcn_wmma_f32_16x16x32_bf16(false, au[kc], false, bu,
                                                                   (short)0, accu, false, false);
                    accw = __builtin_amdgcn_wmma_f32_16x16x32_bf16(false, ah[kc], false, bw,
                                                                   (short)0, accw, false, false);
                }
                int col = tile * 16 + n;
                float bias = wsUb[col];
#pragma unroll
                for (int v = 0; v < 8; ++v) {
                    int m = v + kh * 8;              // C layout: VGPR v -> M=v / v+8
                    gates[m * GH + col] = accu[v] + accw[v] + bias;
                }
            } else {
                int tv = tile - NT_G;
                v16bf bv = *(const v16bf*)(wsV + ((size_t)tv * 32 + lane) * 16);
                v8f acc = {};
                acc = __builtin_amdgcn_wmma_f32_16x16x32_bf16(false, ae, false, bv,
                                                              (short)0, acc, false, false);
                int col = tv * 16 + n;
                float bias = (col < H_) ? Vb[col] : 0.0f;
#pragma unroll
                for (int v = 0; v < 8; ++v) {
                    int m = v + kh * 8;
                    vebuf[m * 208 + col] = acc[v] + bias;   // cols 200..207 = dead padding
                }
            }
        }
        __syncthreads();        // B2: gates/vebuf visible; all Au/Ah reads complete

        // -------- LSTM cell update --------
        for (int idx = tid; idx < 16 * H_; idx += 256) {
            int b = idx / H_, hh = idx - b * H_;
            float4 g4 = *(const float4*)&gates[b * GH + 4 * hh];   // i,f,g,o interleaved
            float ig = 1.0f / (1.0f + __expf(-g4.x));
            float fg = 1.0f / (1.0f + __expf(-g4.y));
            float gg = tanhf(g4.z);
            float og = 1.0f / (1.0f + __expf(-g4.w));
            float c  = fg * cbuf[idx] + ig * gg;
            float h  = og * tanhf(c) + tanhf(vebuf[b * 208 + hh]);
            cbuf[idx] = c;
            Ah[b * AST + hh] = (__bf16)h;                          // next-step A operand
            out[((size_t)(b0 + b) * T_ + t) * H_ + hh] = h;        // [B,T,H]
        }
        // no barrier: next stage() touches only Au/Ae (nothing reads them now);
        // B1 of the next iteration orders Ah writes before the next GEMM.
    }
}

extern "C" void kernel_launch(void* const* d_in, const int* in_sizes, int n_in,
                              void* d_out, int out_size, void* d_ws, size_t ws_size,
                              hipStream_t stream) {
    const float* emb   = (const float*)d_in[0];
    const float* hpast = (const float*)d_in[1];
    const float* Uw    = (const float*)d_in[2];
    const float* Ub    = (const float*)d_in[3];
    const float* Ww    = (const float*)d_in[4];
    const float* Vw    = (const float*)d_in[5];
    const float* Vb    = (const float*)d_in[6];
    const float* h0    = (const float*)d_in[7];
    const float* c0    = (const float*)d_in[8];
    float* out = (float*)d_out;

    // ws layout (bytes): U frags [0,358400) | W frags [358400,716800)
    //                    V frags [716800,730112) | permuted U_b [730112,733312)
    char* ws = (char*)d_ws;
    __bf16* wsU  = (__bf16*)(ws);
    __bf16* wsW  = (__bf16*)(ws + 358400);
    __bf16* wsV  = (__bf16*)(ws + 716800);
    float*  wsUb = (float*) (ws + 730112);

    const int NTOT = 2 * (NT_G * KC * 32 * 16) + NT_V * 512 + GH;  // 365856
    prep_kernel<<<(NTOT + 255) / 256, 256, 0, stream>>>(Uw, Ub, Ww, Vw, wsU, wsW, wsV, wsUb);
    lstm_scan_kernel<<<4, 256, 0, stream>>>(emb, hpast, wsU, wsW, wsV, wsUb, Vb, h0, c0, out);
}
